// QwenFullAttention_34256659153031
// MI455X (gfx1250) — compile-verified, matrix operand order checked by reading, not run on hardware
//
#include <hip/hip_runtime.h>
#include <hip/hip_bf16.h>

typedef __bf16 v16bf __attribute__((ext_vector_type(16)));
typedef float  v8f   __attribute__((ext_vector_type(8)));
typedef int    i32x4 __attribute__((vector_size(16)));   // matches async-LDS builtin pointee
typedef unsigned short u16;
typedef unsigned int   u32;

#define AS1 __attribute__((address_space(1)))
#define AS3 __attribute__((address_space(3)))

// ---------- bf16 helpers (bit-level, round-to-nearest-even) ----------
__device__ __forceinline__ u16 f2bf(float x) {
    u32 u = __float_as_uint(x);
    u32 r = u + 0x7FFFu + ((u >> 16) & 1u);
    return (u16)(r >> 16);
}
__device__ __forceinline__ float bf2f(u16 h) {
    return __uint_as_float(((u32)h) << 16);
}

// ---------- WMMA wrapper ----------
__device__ __forceinline__ v8f wmma_bf16(v16bf a, v16bf b, v8f c) {
    return __builtin_amdgcn_wmma_f32_16x16x32_bf16(false, a, false, b, (short)0, c, false, false);
}

// ---------- async global->LDS copy (GLOBAL_LOAD_ASYNC_TO_LDS_B128, ASYNCcnt) ----------
#if __has_builtin(__builtin_amdgcn_global_load_async_to_lds_b128)
#define HAVE_ASYNC_LDS 1
#else
#define HAVE_ASYNC_LDS 0
#endif

__device__ __forceinline__ void cp16_g2l(u16* l, const u16* g) {
#if HAVE_ASYNC_LDS
    __builtin_amdgcn_global_load_async_to_lds_b128((AS1 i32x4*)g, (AS3 i32x4*)l, 0, 0);
#else
    *(uint4*)l = *(const uint4*)g;
#endif
}
__device__ __forceinline__ void async_wait() {
#if HAVE_ASYNC_LDS
#if __has_builtin(__builtin_amdgcn_s_wait_asynccnt)
    __builtin_amdgcn_s_wait_asynccnt(0);
#else
    asm volatile("s_wait_asynccnt 0" ::: "memory");
#endif
#endif
}

// ---------- fragment gather from an LDS tile ----------
// Tile is row-major [rows][rowStride] of bf16 bit patterns (u16).
// Builds the 16-bit A/B operand layout for V_WMMA_*_16X16X32 (wave32):
//   lanes 0-15 : row = row0+lane,    K = {0..7, 16..23}  of the 32-wide window
//   lanes16-31 : row = row0+lane-16, K = {8..15,24..31}
__device__ __forceinline__ v16bf frag_ld(const u16* p, int rowStride, int row0, int col0, int lane) {
    const int hf = (lane >> 4) & 1;
    const int r  = row0 + (lane & 15);
    const u32* rp = (const u32*)(p + (size_t)r * rowStride + col0);
    union { v16bf v; u32 u[8]; } f;
#pragma unroll
    for (int v = 0; v < 8; ++v)
        f.u[v] = rp[(v & 3) + hf * 4 + ((v >> 2) << 3)];
    return f.v;
}

// ============================================================
// Prep: fp32 -> bf16 conversions
// ============================================================
__global__ void conv_f2bf_kernel(const float* __restrict__ in, u16* __restrict__ out, long n) {
    long i = (long)blockIdx.x * 256 + threadIdx.x;
    if (i < n) out[i] = f2bf(in[i]);
}

__global__ void pack_wqkv_kernel(const float* __restrict__ Wq, const float* __restrict__ Wk,
                                 const float* __restrict__ Wv, u16* __restrict__ out) {
    long i = (long)blockIdx.x * 256 + threadIdx.x;           // over 2048*6144
    if (i >= 2048L * 6144) return;
    int k = (int)(i / 6144), n = (int)(i % 6144);
    float v;
    if (n < 4096)      v = Wq[(size_t)k * 4096 + n];
    else if (n < 5120) v = Wk[(size_t)k * 1024 + (n - 4096)];
    else               v = Wv[(size_t)k * 1024 + (n - 5120)];
    out[i] = f2bf(v);
}

// ============================================================
// Tiled bf16 WMMA GEMM: C[M,N] = A[M,K] * B[K,N]
// Block: 256 threads (8 waves), tile 64(M) x 128(N), K-step 32.
// A panel staged via async global->LDS b128; B panel staged K-transposed.
// ============================================================
template <typename OutT>
__global__ __launch_bounds__(256) void gemm_bf16_kernel(const u16* __restrict__ A,
                                                        const u16* __restrict__ B,
                                                        OutT* __restrict__ C,
                                                        int M, int N, int K) {
    __shared__ u16 As[64 * 32];      // [m][k]
    __shared__ u16 Bs[128 * 32];     // [n][k]  (K-transposed so pairs are adjacent)
    const int t = threadIdx.x;
    const int lane = t & 31, w = t >> 5;
    const int m0 = blockIdx.y * 64;
    const int n0 = blockIdx.x * 128;
    const int wm = (w & 1) * 32;     // wave M offset in tile
    const int wn = (w >> 1) * 32;    // wave N offset in tile

    v8f acc[2][2];
#pragma unroll
    for (int i = 0; i < 2; ++i)
#pragma unroll
        for (int j = 0; j < 2; ++j) acc[i][j] = {};

    for (int k0 = 0; k0 < K; k0 += 32) {
        // A tile 64x32: 256 threads x one 16B async chunk
        {
            int m = t >> 2, kc = (t & 3) * 8;
            cp16_g2l(&As[m * 32 + kc], &A[(size_t)(m0 + m) * K + k0 + kc]);
        }
        // B tile 32x128, stored transposed (scatter -> element-wise)
#pragma unroll
        for (int i = 0; i < 16; ++i) {
            int idx = i * 256 + t;
            int k = idx >> 7, n = idx & 127;
            Bs[n * 32 + k] = B[(size_t)(k0 + k) * N + n0 + n];
        }
        // prefetch next K panels (global_prefetch_b8)
        if (k0 + 32 < K) {
            __builtin_prefetch(&A[(size_t)(m0 + (t >> 2)) * K + k0 + 32 + (t & 3) * 8], 0, 1);
            __builtin_prefetch(&B[(size_t)(k0 + 32 + (t >> 7)) * N + n0 + (t & 127)], 0, 1);
        }
        async_wait();
        __syncthreads();

        v16bf a0 = frag_ld(As, 32, wm + 0,  0, lane);
        v16bf a1 = frag_ld(As, 32, wm + 16, 0, lane);
        v16bf b0 = frag_ld(Bs, 32, wn + 0,  0, lane);
        v16bf b1 = frag_ld(Bs, 32, wn + 16, 0, lane);
        acc[0][0] = wmma_bf16(a0, b0, acc[0][0]);
        acc[0][1] = wmma_bf16(a0, b1, acc[0][1]);
        acc[1][0] = wmma_bf16(a1, b0, acc[1][0]);
        acc[1][1] = wmma_bf16(a1, b1, acc[1][1]);
        __syncthreads();
    }

    const int hf = lane >> 4, nl = lane & 15;
#pragma unroll
    for (int im = 0; im < 2; ++im)
#pragma unroll
        for (int in = 0; in < 2; ++in)
#pragma unroll
            for (int r = 0; r < 8; ++r) {
                int m = m0 + wm + im * 16 + r + 8 * hf;   // C layout: VGPR r -> row r + 8*half
                int n = n0 + wn + in * 16 + nl;
                float v = acc[im][in][r];
                if constexpr (sizeof(OutT) == 4) C[(size_t)m * N + n] = v;
                else                             C[(size_t)m * N + n] = f2bf(v);
            }
}

// ============================================================
// RMSNorm + RoPE (+ fold attention scale into Q). One wave per (b,s,head).
// heads 0..15 -> Q (q_norm, scale), heads 16..23 -> K (k_norm).
// Lane owns d in {l, l+32, l+64, l+96}; rotate_half pairs (d, d+64) stay in-lane.
// ============================================================
__global__ __launch_bounds__(256) void normrope_kernel(const u16* __restrict__ qkv,
                                                       const float* __restrict__ cosb,
                                                       const float* __restrict__ sinb,
                                                       const float* __restrict__ qnw,
                                                       const float* __restrict__ knw,
                                                       u16* __restrict__ qt,
                                                       u16* __restrict__ ktb) {
    const int t = threadIdx.x, lane = t & 31, w = t >> 5;
    long gw = (long)blockIdx.x * 8 + w;
    int b   = (int)(gw / (2048L * 24));
    int rem = (int)(gw % (2048L * 24));
    int s   = rem / 24;
    int hh  = rem % 24;

    const u16* row = qkv + ((size_t)(b * 2048 + s)) * 6144;
    int col; const float* nw;
    if (hh < 16) { col = hh * 256;               nw = qnw; }
    else         { col = 4096 + (hh - 16) * 128; nw = knw; }

    float x0 = bf2f(row[col + lane]);
    float x1 = bf2f(row[col + lane + 32]);
    float x2 = bf2f(row[col + lane + 64]);
    float x3 = bf2f(row[col + lane + 96]);

    float ss = x0 * x0 + x1 * x1 + x2 * x2 + x3 * x3;
#pragma unroll
    for (int off = 1; off < 32; off <<= 1) ss += __shfl_xor(ss, off, 32);
    float rs = rsqrtf(ss * (1.0f / 128.0f) + 1e-6f);

    x0 *= rs * nw[lane];      x1 *= rs * nw[lane + 32];
    x2 *= rs * nw[lane + 64]; x3 *= rs * nw[lane + 96];

    const float* cp = cosb + ((size_t)(b * 2048 + s)) * 128;
    const float* sp = sinb + ((size_t)(b * 2048 + s)) * 128;
    float y0 = x0 * cp[lane]      - x2 * sp[lane];
    float y1 = x1 * cp[lane + 32] - x3 * sp[lane + 32];
    float y2 = x2 * cp[lane + 64] + x0 * sp[lane + 64];
    float y3 = x3 * cp[lane + 96] + x1 * sp[lane + 96];

    u16* outp;
    if (hh < 16) {
        const float sc = 0.08838834764831845f;   // D^-1/2, folded into Q
        y0 *= sc; y1 *= sc; y2 *= sc; y3 *= sc;
        outp = qt + (((size_t)(b * 16 + hh)) * 2048 + s) * 128;
    } else {
        outp = ktb + (((size_t)(b * 8 + (hh - 16))) * 2048 + s) * 128;
    }
    outp[lane]      = f2bf(y0);
    outp[lane + 32] = f2bf(y1);
    outp[lane + 64] = f2bf(y2);
    outp[lane + 96] = f2bf(y3);
}

// ============================================================
// Causal flash attention with WMMA + gate, writes gated bf16 output.
// Grid: (S/128, H, B). Block: 256 threads = 8 waves; wave owns 16 queries.
// Q and K tiles staged via async global->LDS b128.
// ============================================================
__global__ __launch_bounds__(256) void attn_kernel(const u16* __restrict__ qt,
                                                   const u16* __restrict__ ktb,
                                                   const u16* __restrict__ qkv,
                                                   u16* __restrict__ attng) {
    __shared__ u16 Qs[128 * 128];   // [q][d]        32 KB
    __shared__ u16 Ks[32 * 128];    // [key][d]       8 KB
    __shared__ u16 Vs[128 * 32];    // [d][key]       8 KB (key-transposed for PV)
    __shared__ u16 Ps[8 * 16 * 32]; // per-wave P     8 KB

    const int t = threadIdx.x, lane = t & 31, w = t >> 5;
    const int b = blockIdx.z, h = blockIdx.y;
    const int q0 = blockIdx.x * 128;
    const int kvh = h >> 1;                     // G = 2
    const int hf = lane >> 4, nl = lane & 15;

    // stage Q tile (128 queries x 128 dims) via async b128 copies
#pragma unroll
    for (int i = 0; i < 8; ++i) {
        int idx = i * 256 + t;                  // 2048 chunks of 8 bf16
        int sq = idx >> 4, dc = (idx & 15) * 8;
        cp16_g2l(&Qs[sq * 128 + dc],
                 &qt[(((size_t)(b * 16 + h)) * 2048 + q0 + sq) * 128 + dc]);
    }

    float mrow[8], lrow[8];
    v8f o[8];
#pragma unroll
    for (int r = 0; r < 8; ++r) { mrow[r] = -1e30f; lrow[r] = 0.0f; o[r] = {}; }

    const int qw  = q0 + w * 16;                // wave's first query
    const int qhi = qw + 15;
    const int nkt = (q0 + 128) / 32;            // key tiles for this block

    for (int ktile = 0; ktile < nkt; ++ktile) {
        const int k0 = ktile * 32;
        // K tile (32x128) async
#pragma unroll
        for (int i = 0; i < 2; ++i) {
            int idx = i * 256 + t;              // 512 chunks of 8 bf16
            int kk = idx >> 4, dc = (idx & 15) * 8;
            cp16_g2l(&Ks[kk * 128 + dc],
                     &ktb[(((size_t)(b * 8 + kvh)) * 2048 + k0 + kk) * 128 + dc]);
        }
        // V tile key-transposed (scatter -> element-wise)
#pragma unroll
        for (int i = 0; i < 16; ++i) {
            int idx = i * 256 + t;
            int kk = idx >> 7, d = idx & 127;
            Vs[d * 32 + kk] = qkv[((size_t)(b * 2048 + k0 + kk)) * 6144 + 5120 + kvh * 128 + d];
        }
        async_wait();
        __syncthreads();

        if (k0 <= qhi) {                        // wave-uniform causal skip
            v8f s0 = {}, s1 = {};
#pragma unroll
            for (int kc = 0; kc < 4; ++kc) {    // D = 128 in 4 chunks of 32
                v16bf a  = frag_ld(Qs, 128, w * 16, kc * 32, lane);
                v16bf b0 = frag_ld(Ks, 128, 0,      kc * 32, lane);
                v16bf b1 = frag_ld(Ks, 128, 16,     kc * 32, lane);
                s0 = wmma_bf16(a, b0, s0);
                s1 = wmma_bf16(a, b1, s1);
            }
            // causal mask + online softmax (C layout: row = r + 8*hf, col = nl)
#pragma unroll
            for (int r = 0; r < 8; ++r) {
                int qg = qw + r + 8 * hf;
                float v0 = s0[r], v1 = s1[r];
                if (k0 + nl > qg)      v0 = -1e30f;
                if (k0 + 16 + nl > qg) v1 = -1e30f;
                float mx = fmaxf(v0, v1);
#pragma unroll
                for (int off = 1; off < 16; off <<= 1) mx = fmaxf(mx, __shfl_xor(mx, off, 32));
                float mnew = fmaxf(mrow[r], mx);
                float corr = __expf(mrow[r] - mnew);
                float p0 = __expf(v0 - mnew);
                float p1 = __expf(v1 - mnew);
                float sm = p0 + p1;
#pragma unroll
                for (int off = 1; off < 16; off <<= 1) sm += __shfl_xor(sm, off, 32);
                lrow[r] = lrow[r] * corr + sm;
                mrow[r] = mnew;
#pragma unroll
                for (int dt = 0; dt < 8; ++dt) o[dt][r] *= corr;
                int m = r + 8 * hf;
                Ps[(w * 16 + m) * 32 + nl]      = f2bf(p0);
                Ps[(w * 16 + m) * 32 + 16 + nl] = f2bf(p1);
            }
            // per-wave LDS round-trip: C-layout -> A-layout; DS ops in-order per wave
            asm volatile("s_wait_dscnt 0" ::: "memory");
            v16bf pA = frag_ld(Ps + w * 16 * 32, 32, 0, 0, lane);
#pragma unroll
            for (int dt = 0; dt < 8; ++dt) {
                v16bf bV = frag_ld(Vs, 32, dt * 16, 0, lane);
                o[dt] = wmma_bf16(pA, bV, o[dt]);
            }
        }
        __syncthreads();
    }

    // epilogue: normalize, apply sigmoid(gate), store bf16
#pragma unroll
    for (int r = 0; r < 8; ++r) {
        int m = r + 8 * hf;
        int sq = qw + m;
        float inv = 1.0f / lrow[r];
#pragma unroll
        for (int dt = 0; dt < 8; ++dt) {
            int d = dt * 16 + nl;
            float val = o[dt][r] * inv;
            float g   = bf2f(qkv[((size_t)(b * 2048 + sq)) * 6144 + h * 256 + 128 + d]);
            float sig = 1.0f / (1.0f + __expf(-g));
            attng[((size_t)(b * 2048 + sq)) * 2048 + h * 128 + d] = f2bf(val * sig);
        }
    }
}

// ============================================================
// Host-side orchestration
// ============================================================
extern "C" void kernel_launch(void* const* d_in, const int* in_sizes, int n_in,
                              void* d_out, int out_size, void* d_ws, size_t ws_size,
                              hipStream_t stream) {
    const float* hidden = (const float*)d_in[0];   // (B,S,HID)  = (2,2048,2048)
    const float* cosb   = (const float*)d_in[1];   // (B,S,D)
    const float* sinb   = (const float*)d_in[2];
    const float* Wq     = (const float*)d_in[3];   // (2048,4096)
    const float* Wk     = (const float*)d_in[4];   // (2048,1024)
    const float* Wv     = (const float*)d_in[5];   // (2048,1024)
    const float* Wo     = (const float*)d_in[6];   // (2048,2048)
    const float* qnw    = (const float*)d_in[7];
    const float* knw    = (const float*)d_in[8];
    float* out = (float*)d_out;

    // workspace carve-out (~143 MB)
    char* ws = (char*)d_ws;
    u16* Xb    = (u16*)ws; ws += 4096L * 2048 * 2;   // bf16 hidden
    u16* Wqkvb = (u16*)ws; ws += 2048L * 6144 * 2;   // packed [Wq|Wk|Wv]
    u16* Wob   = (u16*)ws; ws += 2048L * 2048 * 2;   // bf16 Wo
    u16* QKV   = (u16*)ws; ws += 4096L * 6144 * 2;   // bf16 QKV projection output
    u16* qtb   = (u16*)ws; ws += 2L * 16 * 2048 * 128 * 2; // roped+scaled Q [b][h][s][d]
    u16* ktb   = (u16*)ws; ws += 2L * 8  * 2048 * 128 * 2; // roped K       [b][kvh][s][d]
    u16* attng = (u16*)ws; ws += 4096L * 2048 * 2;   // gated attention out (bf16)

    // 1) fp32 -> bf16 conversions
    conv_f2bf_kernel<<<(int)((4096L * 2048 + 255) / 256), 256, 0, stream>>>(hidden, Xb, 4096L * 2048);
    pack_wqkv_kernel<<<(int)((2048L * 6144 + 255) / 256), 256, 0, stream>>>(Wq, Wk, Wv, Wqkvb);
    conv_f2bf_kernel<<<(int)((2048L * 2048 + 255) / 256), 256, 0, stream>>>(Wo, Wob, 2048L * 2048);

    // 2) QKV projection: (4096 x 2048) @ (2048 x 6144) -> bf16
    gemm_bf16_kernel<u16><<<dim3(6144 / 128, 4096 / 64), 256, 0, stream>>>(Xb, Wqkvb, QKV, 4096, 6144, 2048);

    // 3) RMSNorm + RoPE for Q and K
    normrope_kernel<<<(2 * 2048 * 24) / 8, 256, 0, stream>>>(QKV, cosb, sinb, qnw, knw, qtb, ktb);

    // 4) causal flash attention + gate
    attn_kernel<<<dim3(2048 / 128, 16, 2), 256, 0, stream>>>(qtb, ktb, QKV, attng);

    // 5) output projection: (4096 x 2048) @ (2048 x 2048) -> fp32 d_out
    gemm_bf16_kernel<float><<<dim3(2048 / 128, 4096 / 64), 256, 0, stream>>>(attng, Wob, out, 4096, 2048, 2048);
}
